// MDGLAAL_58703613002408
// MI455X (gfx1250) — matrix operand order repaired and checked by reading, not compiled, not run on hardware
//
#include <hip/hip_runtime.h>
#include <hip/hip_bf16.h>

typedef __attribute__((ext_vector_type(16))) __bf16 v16bf;
typedef __attribute__((ext_vector_type(8)))  float  v8f;

#define WMMA_BF16(a_, b_, c_) \
  __builtin_amdgcn_wmma_f32_16x16x32_bf16(false, (a_), false, (b_), (short)0, (c_), false, false)

static constexpr int Bc = 4, Tc = 32, Nc = 400, Ec = 128, Hc = 4, Lc = 2;
static constexpr int Gc = Bc * Tc;        // 128 graphs
static constexpr int Mrows = Gc * Nc;     // 51200 rows

// ---------------- WMMA fragment loaders (wave32, bf16 16x16x32) ----------------
// A layout: lane L -> row = L&15, half = L>>4. Elements 0..7 = contiguous K run at
// k0+half*8; elements 8..15 = contiguous K run at k0+16+half*8.
__device__ inline v16bf a_frag_fast(const float* __restrict__ p0) {
  float4 x0 = *(const float4*)(p0);
  float4 x1 = *(const float4*)(p0 + 4);
  float4 y0 = *(const float4*)(p0 + 16);
  float4 y1 = *(const float4*)(p0 + 20);
  v16bf f;
  f[0] = (__bf16)x0.x; f[1] = (__bf16)x0.y; f[2] = (__bf16)x0.z; f[3] = (__bf16)x0.w;
  f[4] = (__bf16)x1.x; f[5] = (__bf16)x1.y; f[6] = (__bf16)x1.z; f[7] = (__bf16)x1.w;
  f[8] = (__bf16)y0.x; f[9] = (__bf16)y0.y; f[10] = (__bf16)y0.z; f[11] = (__bf16)y0.w;
  f[12] = (__bf16)y1.x; f[13] = (__bf16)y1.y; f[14] = (__bf16)y1.z; f[15] = (__bf16)y1.w;
  return f;
}

// A tail: K partially out of range -> address-clamped loads + value select (no branches)
__device__ inline v16bf a_frag_tail(const float* __restrict__ Arow, int k0, int Kb, int half) {
  v16bf f;
#pragma unroll
  for (int e = 0; e < 16; ++e) {
    int kk = k0 + half * 8 + ((e < 8) ? e : (e + 8));
    float x = Arow[(kk < Kb) ? kk : 0];
    f[e] = (__bf16)((kk < Kb) ? x : 0.0f);
  }
  return f;
}

// B layout (contiguous-K case): lane L -> col = L&15, half = L>>4.
// Elements 0..7 = contiguous K run at k0+half*8; elements 8..15 at k0+16+half*8.
__device__ inline v16bf b_frag_fast(const float* __restrict__ p) {
  float4 x0 = *(const float4*)(p);
  float4 x1 = *(const float4*)(p + 4);
  float4 x2 = *(const float4*)(p + 16);
  float4 x3 = *(const float4*)(p + 20);
  v16bf f;
  f[0] = (__bf16)x0.x; f[1] = (__bf16)x0.y; f[2] = (__bf16)x0.z; f[3] = (__bf16)x0.w;
  f[4] = (__bf16)x1.x; f[5] = (__bf16)x1.y; f[6] = (__bf16)x1.z; f[7] = (__bf16)x1.w;
  f[8] = (__bf16)x2.x; f[9] = (__bf16)x2.y; f[10] = (__bf16)x2.z; f[11] = (__bf16)x2.w;
  f[12] = (__bf16)x3.x; f[13] = (__bf16)x3.y; f[14] = (__bf16)x3.z; f[15] = (__bf16)x3.w;
  return f;
}

__device__ inline v16bf b_frag_tail(const float* __restrict__ Bcol, int k0, int Kb, int half) {
  v16bf f;
#pragma unroll
  for (int e = 0; e < 16; ++e) {
    int kk = k0 + half * 8 + ((e < 8) ? e : (e + 8));
    float x = Bcol[(kk < Kb) ? kk : 0];
    f[e] = (__bf16)((kk < Kb) ? x : 0.0f);
  }
  return f;
}

// ---------------- generic GEMM: C = act(A @ W.T + bias [+ resid]) ----------------
// One wave per 16(M) x 64(N) tile; B(k,n) = Bp[n*sn + k] (contiguous in k).
// OOB N columns are address-clamped only (column independence) and never stored.
// act: 0 none, 1 relu, 2 sigmoid.
__global__ void gemm_kernel(const float* __restrict__ A, int lda,
                            const float* __restrict__ Bp, long sn,
                            float* __restrict__ C, int ldc,
                            const float* __restrict__ bias,
                            const float* __restrict__ resid,
                            int Kb, int Nb, int act) {
  int row0 = blockIdx.x * 16, col0 = blockIdx.y * 64;
  int lane = threadIdx.x & 31;
  int half = lane >> 4;
  int m = row0 + (lane & 15);
  const float* Arow = A + (size_t)m * lda;
  const float* Bcol[4];
#pragma unroll
  for (int t = 0; t < 4; ++t) {
    int n = col0 + t * 16 + (lane & 15);
    Bcol[t] = Bp + (size_t)((n < Nb) ? n : (Nb - 1)) * sn;
  }
  v8f acc[4];
#pragma unroll
  for (int t = 0; t < 4; ++t) acc[t] = (v8f){0.f, 0.f, 0.f, 0.f, 0.f, 0.f, 0.f, 0.f};

  int kfull = Kb & ~31;
  for (int k0 = 0; k0 < kfull; k0 += 32) {
    v16bf af = a_frag_fast(Arow + k0 + half * 8);
#pragma unroll
    for (int t = 0; t < 4; ++t) {
      v16bf bf = b_frag_fast(Bcol[t] + k0 + half * 8);
      acc[t] = WMMA_BF16(af, bf, acc[t]);
    }
  }
  if (kfull < Kb) {
    v16bf af = a_frag_tail(Arow, kfull, Kb, half);
#pragma unroll
    for (int t = 0; t < 4; ++t) {
      v16bf bf = b_frag_tail(Bcol[t], kfull, Kb, half);
      acc[t] = WMMA_BF16(af, bf, acc[t]);
    }
  }
#pragma unroll
  for (int t = 0; t < 4; ++t) {
    int n = col0 + t * 16 + (lane & 15);
    if (n >= Nb) continue;
    float bv = bias ? bias[n] : 0.0f;
#pragma unroll
    for (int r = 0; r < 8; ++r) {
      int mm = row0 + half * 8 + r;
      float v = acc[t][r] + bv;
      if (resid) v += resid[(size_t)mm * ldc + n];
      if (act == 1) v = fmaxf(v, 0.0f);
      else if (act == 2) v = 1.0f / (1.0f + expf(-v));
      C[(size_t)mm * ldc + n] = v;
    }
  }
}

// ---------------- tiled transpose: X (G x N x E) -> XT (G x E x N) ----------------
__global__ void transpose_kernel(const float* __restrict__ X, float* __restrict__ XT) {
  __shared__ float tile[16][17];
  int g = blockIdx.z;
  int n0 = blockIdx.x * 16;   // N dim
  int e0 = blockIdx.y * 16;   // E dim
  int tx = threadIdx.x & 15, ty = threadIdx.x >> 4;
  tile[ty][tx] = X[((size_t)g * Nc + n0 + ty) * Ec + e0 + tx];
  __syncthreads();
  XT[((size_t)g * Ec + e0 + ty) * Nc + n0 + tx] = tile[tx][ty];
}

// ---------------- agg = binarize(a1,thr) @ h + eps*h (per graph) ----------------
// A = on-the-fly binarized adjacency (rows contiguous); B from hT (contiguous in K).
__global__ void agg_kernel(const float* __restrict__ a1, const float* __restrict__ thr,
                           const float* __restrict__ h, const float* __restrict__ hT,
                           const float* __restrict__ eps_p, float* __restrict__ out) {
  int g = blockIdx.z;
  int row0 = blockIdx.x * 16, col0 = blockIdx.y * 64;
  float tg = thr[g];
  int lane = threadIdx.x & 31;
  int half = lane >> 4;
  int m = row0 + (lane & 15);
  const float* Arow = a1 + ((size_t)g * Nc + m) * Nc;
  const float* hg = h + (size_t)g * Nc * Ec;
  const float* Bcol[4];
#pragma unroll
  for (int t = 0; t < 4; ++t) {
    int n = col0 + t * 16 + (lane & 15);
    Bcol[t] = hT + ((size_t)g * Ec + n) * Nc;
  }
  v8f acc[4];
#pragma unroll
  for (int t = 0; t < 4; ++t) acc[t] = (v8f){0.f, 0.f, 0.f, 0.f, 0.f, 0.f, 0.f, 0.f};

  const int kfull = Nc & ~31;   // 384
  for (int k0 = 0; k0 < kfull; k0 += 32) {
    const float* p0 = Arow + k0 + half * 8;
    float4 x0 = *(const float4*)(p0);
    float4 x1 = *(const float4*)(p0 + 4);
    float4 y0 = *(const float4*)(p0 + 16);
    float4 y1 = *(const float4*)(p0 + 20);
    v16bf af;
    af[0] = (__bf16)((x0.x > tg) ? 1.f : 0.f); af[1] = (__bf16)((x0.y > tg) ? 1.f : 0.f);
    af[2] = (__bf16)((x0.z > tg) ? 1.f : 0.f); af[3] = (__bf16)((x0.w > tg) ? 1.f : 0.f);
    af[4] = (__bf16)((x1.x > tg) ? 1.f : 0.f); af[5] = (__bf16)((x1.y > tg) ? 1.f : 0.f);
    af[6] = (__bf16)((x1.z > tg) ? 1.f : 0.f); af[7] = (__bf16)((x1.w > tg) ? 1.f : 0.f);
    af[8] = (__bf16)((y0.x > tg) ? 1.f : 0.f); af[9] = (__bf16)((y0.y > tg) ? 1.f : 0.f);
    af[10] = (__bf16)((y0.z > tg) ? 1.f : 0.f); af[11] = (__bf16)((y0.w > tg) ? 1.f : 0.f);
    af[12] = (__bf16)((y1.x > tg) ? 1.f : 0.f); af[13] = (__bf16)((y1.y > tg) ? 1.f : 0.f);
    af[14] = (__bf16)((y1.z > tg) ? 1.f : 0.f); af[15] = (__bf16)((y1.w > tg) ? 1.f : 0.f);
#pragma unroll
    for (int t = 0; t < 4; ++t) {
      v16bf bf = b_frag_fast(Bcol[t] + k0 + half * 8);
      acc[t] = WMMA_BF16(af, bf, acc[t]);
    }
  }
  {  // K tail: 384..399 valid, 400..415 zero (select, no branches)
    int k0 = kfull;
    v16bf af;
#pragma unroll
    for (int e = 0; e < 16; ++e) {
      int kk = k0 + half * 8 + ((e < 8) ? e : (e + 8));
      float x = Arow[(kk < Nc) ? kk : 0];
      af[e] = (__bf16)(((kk < Nc) && (x > tg)) ? 1.f : 0.f);
    }
#pragma unroll
    for (int t = 0; t < 4; ++t) {
      v16bf bf = b_frag_tail(Bcol[t], k0, Nc, half);
      acc[t] = WMMA_BF16(af, bf, acc[t]);
    }
  }
  float ev = eps_p[0];
#pragma unroll
  for (int t = 0; t < 4; ++t) {
    int n = col0 + t * 16 + (lane & 15);
#pragma unroll
    for (int r = 0; r < 8; ++r) {
      int mm = row0 + half * 8 + r;
      out[((size_t)g * Nc + mm) * Ec + n] = acc[t][r] + ev * hg[(size_t)mm * Ec + n];
    }
  }
}

// ---------------- ortho: Gram rowmax + triu squared-sum (WMMA, K=128 exact) ----------------
__global__ void ortho_rowmax_kernel(const float* __restrict__ h, float* __restrict__ rowmax) {
  int g = blockIdx.x, row0 = blockIdx.y * 16;
  const float* ol = h + (size_t)g * Nc * Ec;
  int lane = threadIdx.x & 31, half = lane >> 4;
  int m = row0 + (lane & 15);
  v16bf afr[4];
#pragma unroll
  for (int kc = 0; kc < 4; ++kc) afr[kc] = a_frag_fast(ol + (size_t)m * Ec + kc * 32 + half * 8);
  float rmax[8];
#pragma unroll
  for (int r = 0; r < 8; ++r) rmax[r] = -3.4e38f;
  for (int j = 0; j < Nc / 16; ++j) {
    int n = j * 16 + (lane & 15);
    v8f acc = {0.f, 0.f, 0.f, 0.f, 0.f, 0.f, 0.f, 0.f};
#pragma unroll
    for (int kc = 0; kc < 4; ++kc) {
      v16bf bf = b_frag_fast(ol + (size_t)n * Ec + kc * 32 + half * 8);  // B(k,m)=ol[m][k]
      acc = WMMA_BF16(afr[kc], bf, acc);
    }
#pragma unroll
    for (int r = 0; r < 8; ++r) rmax[r] = fmaxf(rmax[r], acc[r]);
  }
#pragma unroll
  for (int off = 1; off < 16; off <<= 1)
#pragma unroll
    for (int r = 0; r < 8; ++r) rmax[r] = fmaxf(rmax[r], __shfl_xor(rmax[r], off, 32));
  if ((lane & 15) == 0)
#pragma unroll
    for (int r = 0; r < 8; ++r) rowmax[(size_t)g * Nc + row0 + half * 8 + r] = rmax[r];
}

__global__ void ortho_sum_kernel(const float* __restrict__ h, const float* __restrict__ rowmax,
                                 float* __restrict__ opart) {
  int g = blockIdx.x, row0 = blockIdx.y * 16;
  const float* ol = h + (size_t)g * Nc * Ec;
  int lane = threadIdx.x & 31, half = lane >> 4;
  int m = row0 + (lane & 15);
  v16bf afr[4];
#pragma unroll
  for (int kc = 0; kc < 4; ++kc) afr[kc] = a_frag_fast(ol + (size_t)m * Ec + kc * 32 + half * 8);
  float rmx[8];
#pragma unroll
  for (int r = 0; r < 8; ++r) rmx[r] = rowmax[(size_t)g * Nc + row0 + half * 8 + r];
  float lsum = 0.f;
  for (int j = 0; j < Nc / 16; ++j) {
    int n = j * 16 + (lane & 15);
    v8f acc = {0.f, 0.f, 0.f, 0.f, 0.f, 0.f, 0.f, 0.f};
#pragma unroll
    for (int kc = 0; kc < 4; ++kc) {
      v16bf bf = b_frag_fast(ol + (size_t)n * Ec + kc * 32 + half * 8);
      acc = WMMA_BF16(afr[kc], bf, acc);
    }
#pragma unroll
    for (int r = 0; r < 8; ++r) {
      int nrow = row0 + half * 8 + r;
      int mcol = j * 16 + (lane & 15);
      if (mcol >= nrow) {
        float val = acc[r] / rmx[r] - ((mcol == nrow) ? 1.0f : 0.0f);
        lsum += val * val;
      }
    }
  }
#pragma unroll
  for (int off = 1; off < 32; off <<= 1) lsum += __shfl_xor(lsum, off, 32);
  if (lane == 0) opart[(size_t)g * 25 + blockIdx.y] = lsum;
}

__global__ void ortho_fin_kernel(const float* __restrict__ opart, float* __restrict__ out_reg) {
  __shared__ float sh[128];
  int g = threadIdx.x;
  float s = 0.f;
  for (int j = 0; j < 25; ++j) s += opart[(size_t)g * 25 + j];
  sh[g] = sqrtf(s);
  __syncthreads();
  for (int off = 64; off > 0; off >>= 1) { if (g < off) sh[g] += sh[g + off]; __syncthreads(); }
  if (g == 0) out_reg[0] += sh[0] * (1.0f / 128.0f);
}

// ---------------- exact quantile via 4-pass radix select ----------------
__global__ void quantile_kernel(const float* __restrict__ a1, float* __restrict__ thr) {
  int g = blockIdx.x;
  int tid = threadIdx.x;
  const float* data = a1 + (size_t)g * Nc * Nc;
  const int Mtot = Nc * Nc;
  float fq = 0.7f * (float)(Mtot - 1);
  int lo = (int)fq;
  float frac = fq - (float)lo;
  __shared__ unsigned int histo[256];
  __shared__ unsigned int sel_prefix;
  __shared__ int sel_rank;
  float vals[2];
  for (int which = 0; which < 2; ++which) {
    int rank = lo + which;
    unsigned prefix = 0u;
    for (int pass = 0; pass < 4; ++pass) {
      int shift = 24 - pass * 8;
      for (int i = tid; i < 256; i += blockDim.x) histo[i] = 0u;
      __syncthreads();
      unsigned mask = (pass == 0) ? 0u : (0xFFFFFFFFu << (32 - 8 * pass));
      for (int i = tid; i < Mtot; i += blockDim.x) {
        unsigned bits = __float_as_uint(data[i]);
        unsigned key = bits ^ ((bits >> 31) ? 0xFFFFFFFFu : 0x80000000u);
        if ((key & mask) == prefix) atomicAdd(&histo[(key >> shift) & 0xFFu], 1u);
      }
      __syncthreads();
      if (tid == 0) {
        unsigned cum = 0; int b = 0;
        for (; b < 256; ++b) { if (cum + histo[b] > (unsigned)rank) break; cum += histo[b]; }
        sel_prefix = prefix | ((unsigned)b << shift);
        sel_rank = rank - (int)cum;
      }
      __syncthreads();
      prefix = sel_prefix;
      rank = sel_rank;
      __syncthreads();
    }
    unsigned bits = prefix ^ ((prefix >> 31) ? 0x80000000u : 0xFFFFFFFFu);
    vals[which] = __uint_as_float(bits);
  }
  if (tid == 0) thr[g] = vals[0] + frac * (vals[1] - vals[0]);
}

// ---------------- BatchNorm (training-mode batch stats), deterministic ----------------
__global__ void colstats_kernel(const float* __restrict__ X, int rows, int cols,
                                float* __restrict__ part) {
  __shared__ float ls[512];
  int tid = threadIdx.x;
  for (int i = tid; i < 2 * cols; i += blockDim.x) ls[i] = 0.f;
  __syncthreads();
  int r0 = blockIdx.x * 256;
  int nr = (r0 + 256 <= rows) ? 256 : (rows - r0);
  const float* Xb = X + (size_t)r0 * cols;
  int total = nr * cols;
  for (int i = tid; i < total; i += blockDim.x) {
    float v = Xb[i];
    int c = i % cols;
    atomicAdd(&ls[c], v);
    atomicAdd(&ls[cols + c], v * v);
  }
  __syncthreads();
  for (int i = tid; i < 2 * cols; i += blockDim.x)
    part[(size_t)blockIdx.x * 2 * cols + i] = ls[i];
}

__global__ void bnfinal_kernel(const float* __restrict__ part, int nblocks, int rows, int cols,
                               const float* __restrict__ g, const float* __restrict__ b,
                               float* __restrict__ scale, float* __restrict__ shift) {
  int c = threadIdx.x;
  if (c >= cols) return;
  float s = 0.f, sq = 0.f;
  for (int k = 0; k < nblocks; ++k) {
    s += part[(size_t)k * 2 * cols + c];
    sq += part[(size_t)k * 2 * cols + cols + c];
  }
  float mu = s / (float)rows;
  float var = sq / (float)rows - mu * mu;
  float a = g[c] * rsqrtf(var + 1e-5f);
  scale[c] = a;
  shift[c] = b[c] - mu * a;
}

// act: 1 relu, 3 exact gelu
__global__ void bnapply_kernel(const float* __restrict__ X, float* __restrict__ Y,
                               const float* __restrict__ scale, const float* __restrict__ shift,
                               size_t total, int cols, int act) {
  size_t i = (size_t)blockIdx.x * blockDim.x + threadIdx.x;
  if (i >= total) return;
  int c = (int)(i % cols);
  float v = X[i] * scale[c] + shift[c];
  if (act == 1) v = fmaxf(v, 0.0f);
  else if (act == 3) v = 0.5f * v * (1.0f + erff(v * 0.70710678f));
  Y[i] = v;
}

// ---------------- LayerNorm (E=128), optional residual ----------------
__global__ void ln_kernel(const float* __restrict__ X, const float* __restrict__ R,
                          const float* __restrict__ g, const float* __restrict__ b,
                          float* __restrict__ Y) {
  int row = blockIdx.x, c = threadIdx.x;
  __shared__ float sh[128];
  float v = X[(size_t)row * Ec + c] + (R ? R[(size_t)row * Ec + c] : 0.f);
  sh[c] = v; __syncthreads();
  for (int off = 64; off > 0; off >>= 1) { if (c < off) sh[c] += sh[c + off]; __syncthreads(); }
  float mu = sh[0] * (1.f / Ec);
  __syncthreads();
  float d = v - mu;
  sh[c] = d * d; __syncthreads();
  for (int off = 64; off > 0; off >>= 1) { if (c < off) sh[c] += sh[c + off]; __syncthreads(); }
  float var = sh[0] * (1.f / Ec);
  Y[(size_t)row * Ec + c] = d * rsqrtf(var + 1e-5f) * g[c] + b[c];
}

// ---------------- SERO readouts ----------------
__global__ void xr_kernel(const float* __restrict__ h, float* __restrict__ xr) {
  int row = blockIdx.x;                 // row = t*B + b
  int t = row / Bc, b = row % Bc;
  int e = threadIdx.x;
  const float* hp = h + ((size_t)(b * Tc + t) * Nc) * Ec + e;
  float s = 0.f;
  for (int n = 0; n < Nc; ++n) s += hp[(size_t)n * Ec];
  xr[(size_t)row * Ec + e] = s * (1.0f / Nc);
}

__global__ void hro_kernel(const float* __restrict__ h, const float* __restrict__ attnb,
                           float* __restrict__ hro) {
  int row = blockIdx.x;
  int t = row / Bc, b = row % Bc;
  int e = threadIdx.x;
  const float* hp = h + ((size_t)(b * Tc + t) * Nc) * Ec + e;
  const float* ap = attnb + (size_t)row * Nc;
  float s = 0.f;
  for (int n = 0; n < Nc; ++n) s += hp[(size_t)n * Ec] * ap[n];
  hro[(size_t)row * Ec + e] = s * (1.0f / Nc);
}

// ---------------- tiny MHA (S=32, dh=32), one block per (b,h) ----------------
__global__ void mha_kernel(const float* __restrict__ qkv, float* __restrict__ obuf) {
  int bh = blockIdx.x;
  int b = bh / Hc, h = bh % Hc;
  int tid = threadIdx.x;
  __shared__ float qsh[1024], ksh[1024], vsh[1024], sc[1024];
  for (int i = tid; i < 1024; i += blockDim.x) {
    int t = i >> 5, d = i & 31;
    size_t base = (size_t)(t * Bc + b) * 384 + h * 32 + d;
    qsh[i] = qkv[base];
    ksh[i] = qkv[base + 128];
    vsh[i] = qkv[base + 256];
  }
  __syncthreads();
  for (int i = tid; i < 1024; i += blockDim.x) {
    int s = i >> 5, t = i & 31;
    float a = 0.f;
    for (int d = 0; d < 32; ++d) a += qsh[s * 32 + d] * ksh[t * 32 + d];
    sc[i] = a * 0.17677669529f;   // 1/sqrt(32)
  }
  __syncthreads();
  if (tid < 32) {
    float mx = -3.4e38f;
    for (int t = 0; t < 32; ++t) mx = fmaxf(mx, sc[tid * 32 + t]);
    float sum = 0.f;
    for (int t = 0; t < 32; ++t) { float e = expf(sc[tid * 32 + t] - mx); sc[tid * 32 + t] = e; sum += e; }
    float inv = 1.0f / sum;
    for (int t = 0; t < 32; ++t) sc[tid * 32 + t] *= inv;
  }
  __syncthreads();
  for (int i = tid; i < 1024; i += blockDim.x) {
    int s = i >> 5, d = i & 31;
    float a = 0.f;
    for (int t = 0; t < 32; ++t) a += sc[s * 32 + t] * vsh[t * 32 + d];
    obuf[(size_t)(s * Bc + b) * Ec + h * 32 + d] = a;
  }
}

// ---------------- latent sum over sequence ----------------
__global__ void latent_kernel(const float* __restrict__ xa2, float* __restrict__ out, int l) {
  int b = blockIdx.x, e = threadIdx.x;
  float s = 0.f;
  for (int t = 0; t < Tc; ++t) s += xa2[(size_t)(t * Bc + b) * Ec + e];
  out[(size_t)b * (Lc * Ec) + l * Ec + e] = s;
}

__global__ void zero_kernel(float* __restrict__ p, int n) {
  int i = blockIdx.x * blockDim.x + threadIdx.x;
  if (i < n) p[i] = 0.f;
}

// ---------------- host orchestration ----------------
extern "C" void kernel_launch(void* const* d_in, const int* in_sizes, int n_in,
                              void* d_out, int out_size, void* d_ws, size_t ws_size,
                              hipStream_t stream) {
  const float* v1 = (const float*)d_in[0];
  const float* a1 = (const float*)d_in[1];
  const float* init_b = (const float*)d_in[4];
  const float* init_w = (const float*)d_in[5];
  // layer leaves sorted: 0 b_o, 1 b_qkv, 2 gin_b1, 3 gin_b2, 4 gin_bb1, 5 gin_bb2,
  // 6 gin_eps, 7 gin_g1, 8 gin_g2, 9 gin_w1, 10 gin_w2, 11 ln1_b, 12 ln1_g,
  // 13 ln2_b, 14 ln2_g, 15 mlp_b1, 16 mlp_b2, 17 mlp_w1, 18 mlp_w2, 19 sero_ba,
  // 20 sero_bb, 21 sero_be, 22 sero_g, 23 sero_wa, 24 sero_we, 25 w_o, 26 w_qkv
  auto LP = [&](int l, int idx) { return (const float*)d_in[6 + l * 27 + idx]; };

  float* ws = (float*)d_ws;
  float* thr    = ws;                    // 128
  float* stats  = thr + 128;             // 51200 (200 blocks * 256)
  float* scale  = stats + 51200;         // 256
  float* shift  = scale + 256;           // 256
  float* xr     = shift + 256;           // 16384
  float* emb    = xr + 16384;            // 16384
  float* attnb  = emb + 16384;           // 51200
  float* hro    = attnb + 51200;         // 16384
  float* qkvb   = hro + 16384;           // 49152
  float* obuf   = qkvb + 49152;          // 16384
  float* mo     = obuf + 16384;          // 16384
  float* xab    = mo + 16384;            // 16384
  float* mh1    = xab + 16384;           // 32768
  float* xa2    = mh1 + 32768;           // 16384
  float* rowmax = xa2 + 16384;           // 51200
  float* opart  = rowmax + 51200;        // 3200
  float* hbuf   = opart + 3200;          // 6,553,600
  float* aggb   = hbuf + (size_t)Mrows * Ec;   // 6,553,600
  float* htb    = aggb + (size_t)Mrows * Ec;   // 6,553,600 (h transposed per graph)

  float* outp = (float*)d_out;
  dim3 w32(32);
  const int nblk = (Mrows + 255) / 256;  // 200

  zero_kernel<<<(1025 + 255) / 256, 256, 0, stream>>>(outp, 1025);
  quantile_kernel<<<Gc, 256, 0, stream>>>(a1, thr);

  // h = v1 @ init_w.T + init_b  (init_w is (E,N): B(k,n)=init_w[n*N+k], contig k)
  gemm_kernel<<<dim3(Mrows / 16, 2), w32, 0, stream>>>(
      v1, Nc, init_w, Nc, hbuf, Ec, init_b, nullptr, Nc, Ec, 0);

  for (int l = 0; l < Lc; ++l) {
    // transpose h per graph so agg's B operand is contiguous in K
    transpose_kernel<<<dim3(Nc / 16, Ec / 16, Gc), 256, 0, stream>>>(hbuf, htb);
    // GIN aggregation: agg = adj@h + eps*h
    agg_kernel<<<dim3(Nc / 16, 2, Gc), w32, 0, stream>>>(a1, thr, hbuf, htb, LP(l, 6), aggb);

    // GIN MLP layer 1: agg @ gin_w1.T + b1 -> hbuf ; BN ; relu
    gemm_kernel<<<dim3(Mrows / 16, 2), w32, 0, stream>>>(
        aggb, Ec, LP(l, 9), Ec, hbuf, Ec, LP(l, 2), nullptr, Ec, Ec, 0);
    colstats_kernel<<<nblk, 256, 0, stream>>>(hbuf, Mrows, Ec, stats);
    bnfinal_kernel<<<1, 128, 0, stream>>>(stats, nblk, Mrows, Ec, LP(l, 7), LP(l, 4), scale, shift);
    bnapply_kernel<<<(int)(((size_t)Mrows * Ec + 255) / 256), 256, 0, stream>>>(
        hbuf, hbuf, scale, shift, (size_t)Mrows * Ec, Ec, 1);

    // GIN MLP layer 2 -> aggb ; BN ; relu -> hbuf
    gemm_kernel<<<dim3(Mrows / 16, 2), w32, 0, stream>>>(
        hbuf, Ec, LP(l, 10), Ec, aggb, Ec, LP(l, 3), nullptr, Ec, Ec, 0);
    colstats_kernel<<<nblk, 256, 0, stream>>>(aggb, Mrows, Ec, stats);
    bnfinal_kernel<<<1, 128, 0, stream>>>(stats, nblk, Mrows, Ec, LP(l, 8), LP(l, 5), scale, shift);
    bnapply_kernel<<<(int)(((size_t)Mrows * Ec + 255) / 256), 256, 0, stream>>>(
        aggb, hbuf, scale, shift, (size_t)Mrows * Ec, Ec, 1);

    // SERO readout
    xr_kernel<<<Gc, 128, 0, stream>>>(hbuf, xr);
    gemm_kernel<<<dim3(8, 2), w32, 0, stream>>>(
        xr, Ec, LP(l, 24), Ec, emb, Ec, LP(l, 21), nullptr, Ec, Ec, 0);
    colstats_kernel<<<1, 256, 0, stream>>>(emb, Gc, Ec, stats);
    bnfinal_kernel<<<1, 128, 0, stream>>>(stats, 1, Gc, Ec, LP(l, 22), LP(l, 20), scale, shift);
    bnapply_kernel<<<(Gc * Ec + 255) / 256, 256, 0, stream>>>(
        emb, emb, scale, shift, (size_t)Gc * Ec, Ec, 3);
    gemm_kernel<<<dim3(8, 7), w32, 0, stream>>>(
        emb, Ec, LP(l, 23), Ec, attnb, Nc, LP(l, 19), nullptr, Ec, Nc, 2);
    hro_kernel<<<Gc, 128, 0, stream>>>(hbuf, attnb, hro);

    // Transformer block
    gemm_kernel<<<dim3(8, 6), w32, 0, stream>>>(
        hro, Ec, LP(l, 26), Ec, qkvb, 384, LP(l, 1), nullptr, Ec, 384, 0);
    mha_kernel<<<Bc * Hc, 256, 0, stream>>>(qkvb, obuf);
    gemm_kernel<<<dim3(8, 2), w32, 0, stream>>>(
        obuf, Ec, LP(l, 25), Ec, mo, Ec, LP(l, 0), nullptr, Ec, Ec, 0);
    ln_kernel<<<Gc, 128, 0, stream>>>(mo, nullptr, LP(l, 12), LP(l, 11), xab);
    gemm_kernel<<<dim3(8, 4), w32, 0, stream>>>(
        xab, Ec, LP(l, 17), Ec, mh1, 256, LP(l, 15), nullptr, Ec, 256, 1);
    gemm_kernel<<<dim3(8, 2), w32, 0, stream>>>(
        mh1, 256, LP(l, 18), 256, xa2, Ec, LP(l, 16), nullptr, 256, Ec, 0);
    ln_kernel<<<Gc, 128, 0, stream>>>(xa2, xab, LP(l, 14), LP(l, 13), xa2);
    latent_kernel<<<Bc, 128, 0, stream>>>(xa2, outp, l);

    // orthogonality regularizer
    ortho_rowmax_kernel<<<dim3(Gc, 25), w32, 0, stream>>>(hbuf, rowmax);
    ortho_sum_kernel<<<dim3(Gc, 25), w32, 0, stream>>>(hbuf, rowmax, opart);
    ortho_fin_kernel<<<1, 128, 0, stream>>>(opart, outp + 1024);
  }
}
